// GraphAttention_48515950576492
// MI455X (gfx1250) — matrix operand order repaired and checked by reading, not compiled, not run on hardware
//
#include <hip/hip_runtime.h>
#include <hip/hip_bf16.h>

#define N_NODES 8192
#define IN_DIM  256
#define OUT_DIM 128
#define ALPHA   0.2f
#define LOG2E   1.4426950408889634f

typedef __bf16 bf16_t;
typedef bf16_t v16bf __attribute__((ext_vector_type(16)));
typedef float  v8f   __attribute__((ext_vector_type(8)));

union ABu { uint4 u[2]; v16bf v; };

__device__ __forceinline__ unsigned short f2bf(float f) {
    unsigned int u = __float_as_uint(f);
    unsigned int r = u + 0x7FFFu + ((u >> 16) & 1u);   // round-to-nearest-even bf16
    return (unsigned short)(r >> 16);
}

// ---------------------------------------------------------------------------
// Kernel A: Wh = x @ W  (fp32 for f1/f2, plus bf16 pre-swizzled into a
// 32x16 B-fragment layout for v_wmma_f32_16x16x32_bf16, k contiguous per lane half).
// ---------------------------------------------------------------------------
__global__ void __launch_bounds__(128)
wh_kernel(const float* __restrict__ x, const float* __restrict__ W,
          float* __restrict__ Wh, unsigned short* __restrict__ WhB) {
    __shared__ float xrow[IN_DIM];
    const int i = blockIdx.x;
    const int n = threadIdx.x;
    xrow[n]       = x[(size_t)i * IN_DIM + n];
    xrow[n + 128] = x[(size_t)i * IN_DIM + n + 128];
    __syncthreads();
    float acc = 0.f;
#pragma unroll 8
    for (int k = 0; k < IN_DIM; k++)
        acc = fmaf(xrow[k], W[(size_t)k * OUT_DIM + n], acc);
    Wh[(size_t)i * OUT_DIM + n] = acc;

    // B-fragment swizzle: fragment(jb, t, lane L) holds Wh[jb*32 + koff(L) + k][t*16 + (L&15)]
    // for k = 0..15, koff = (L<16)?0:16. Lane loads become contiguous 32B.
    const int jb = i >> 5, jr = i & 31;
    const int k  = jr & 15;
    const int L  = ((jr >> 4) << 4) | (n & 15);
    const int t  = n >> 4;
    const size_t off = (((size_t)jb * 8 + t) * 32 + L) * 16 + k;
    WhB[off] = f2bf(acc);
}

// ---------------------------------------------------------------------------
// Kernel B: f1 = Wh@a1, f2 = Wh@a2
// ---------------------------------------------------------------------------
__global__ void __launch_bounds__(256)
f_kernel(const float* __restrict__ Wh, const float* __restrict__ a,
         float* __restrict__ f1, float* __restrict__ f2) {
    const int i = blockIdx.x * 256 + threadIdx.x;
    float s1 = 0.f, s2 = 0.f;
#pragma unroll 8
    for (int n = 0; n < OUT_DIM; n++) {
        float w = Wh[(size_t)i * OUT_DIM + n];
        s1 = fmaf(w, a[n], s1);
        s2 = fmaf(w, a[OUT_DIM + n], s2);
    }
    f1[i] = s1;
    f2[i] = s2;
}

// ---------------------------------------------------------------------------
// Kernel C: maxf2 = max_j f2[j]  (per-row softmax shift C_i = leaky(f1_i+maxf2);
// softmax is shift-invariant so any row upper bound is exact).
// ---------------------------------------------------------------------------
__global__ void __launch_bounds__(256)
maxf2_kernel(const float* __restrict__ f2, float* __restrict__ mf2) {
    __shared__ float sm[256];
    float m = -1e30f;
    for (int i = threadIdx.x; i < N_NODES; i += 256) m = fmaxf(m, f2[i]);
    sm[threadIdx.x] = m;
    __syncthreads();
    for (int s = 128; s > 0; s >>= 1) {
        if (threadIdx.x < s) sm[threadIdx.x] = fmaxf(sm[threadIdx.x], sm[threadIdx.x + s]);
        __syncthreads();
    }
    if (threadIdx.x == 0) mf2[0] = sm[0];
}

// ---------------------------------------------------------------------------
// Kernel D: fused masked-softmax attention + (attention @ Wh).
// 512 blocks x 128 threads (4 waves). Per j-block of 32:
//   - 16 coalesced non-temporal adj loads (one clause, consumed branchlessly)
//   - 16 exp2 -> bf16 P tile staged in per-wave LDS slab (80B pitch)
//   - 2 ds_load_b128 A-fragment reads (k contiguous per lane half, same order as B)
//   - 16 global_load_b128 B fragments preloaded into distinct regs (one clause)
//   - 9x v_wmma_f32_16x16x32_bf16 (8 output tiles + ones-column row sums)
// ---------------------------------------------------------------------------
__global__ void __launch_bounds__(128)
gat_kernel(const int* __restrict__ adj, const unsigned short* __restrict__ WhB,
           const float* __restrict__ f1, const float* __restrict__ f2,
           const float* __restrict__ mf2p, float* __restrict__ out) {
    __shared__ __align__(16) unsigned short ptile[4 * 16 * 40];
    __shared__ float red[4][16][OUT_DIM];
    __shared__ float redd[4][16];

    const int tid  = threadIdx.x;
    const int w    = tid >> 5;
    const int lane = tid & 31;
    const int i0   = blockIdx.x * 16;

    unsigned short* tile = &ptile[w * 16 * 40];
    const int M     = lane & 15;                 // A-fragment row for this lane
    const int kbyte = (lane < 16) ? 0 : 32;      // K-half byte offset within row

    const float mf2 = mf2p[0];
    float f1r[16], cexp[16];
#pragma unroll
    for (int r = 0; r < 16; r++) {
        float v = f1[i0 + r];
        f1r[r]  = v;
        float s = v + mf2;
        s       = fmaxf(s, ALPHA * s);           // leaky(f1_i + max f2): row upper bound
        cexp[r] = -s * LOG2E;                    // exp argument offset, pre-scaled
    }

    const v8f vzero = {0.f, 0.f, 0.f, 0.f, 0.f, 0.f, 0.f, 0.f};
    v8f acc[8];
#pragma unroll
    for (int t = 0; t < 8; t++) acc[t] = vzero;
    v8f dacc = vzero;

    ABu ones;
    ones.u[0] = make_uint4(0x3F803F80u, 0x3F803F80u, 0x3F803F80u, 0x3F803F80u);
    ones.u[1] = ones.u[0];

    const int* arow_base = adj + (size_t)i0 * N_NODES;

    for (int jb = w; jb < N_NODES / 32; jb += 4) {
        const int j0 = jb * 32;
        const int* arow = arow_base + j0 + lane;

        // 1) issue all adj loads as one clause (streamed once -> non-temporal)
        int av[16];
#pragma unroll
        for (int r = 0; r < 16; r++)
            av[r] = __builtin_nontemporal_load(arow + (size_t)r * N_NODES);

        const float f2j = f2[j0 + lane];         // this lane owns column j0+lane

        // 2) branchless p = mask * exp2(leaky(f1+f2)*log2e + c)
#pragma unroll
        for (int r = 0; r < 16; r++) {
            float e   = f1r[r] + f2j;
            e         = fmaxf(e, ALPHA * e);                 // leaky_relu, branchless
            float sel = av[r] > 0 ? 1.0f : 0.0f;             // v_cndmask
            float p   = sel * __builtin_amdgcn_exp2f(fmaf(e, LOG2E, cexp[r]));
            tile[r * 40 + lane] = f2bf(p);
        }

        // 3) A-fragment: two b128 LDS reads per lane (k = 16*(lane/16) + element, same order as B)
        ABu af;
        const char* rowp = (const char*)(tile + M * 40);
        af.u[0] = *(const uint4*)(rowp + kbyte);             // K = kbase .. kbase+7
        af.u[1] = *(const uint4*)(rowp + kbyte + 16);        // K = kbase+8 .. kbase+15

        // 4) preload ALL B fragments into distinct registers (one 16-load clause,
        //    lets the scheduler overlap L2 latency with the WMMA chain)
        ABu bfr[8];
        const unsigned short* wb = WhB + (size_t)jb * 8 * 32 * 16;
#pragma unroll
        for (int t = 0; t < 8; t++) {
            const unsigned short* bp = wb + ((size_t)t * 32 + lane) * 16;
            bfr[t].u[0] = *(const uint4*)bp;                 // contiguous 32B/lane
            bfr[t].u[1] = *(const uint4*)(bp + 8);
        }

        // 5) 9 WMMAs back-to-back
        dacc = __builtin_amdgcn_wmma_f32_16x16x32_bf16(
                   false, af.v, false, ones.v, (short)0, dacc, false, false);
#pragma unroll
        for (int t = 0; t < 8; t++)
            acc[t] = __builtin_amdgcn_wmma_f32_16x16x32_bf16(
                         false, af.v, false, bfr[t].v, (short)0, acc[t], false, false);
    }

    // Cross-wave reduction. C/D layout: VGPR g holds M = g + 8*(lane>=16), N = lane&15.
#pragma unroll
    for (int t = 0; t < 8; t++) {
#pragma unroll
        for (int g = 0; g < 8; g++) {
            int Mr = g + ((lane >> 4) << 3);
            red[w][Mr][t * 16 + (lane & 15)] = acc[t][g];
        }
    }
    if ((lane & 15) == 0) {
#pragma unroll
        for (int g = 0; g < 8; g++) redd[w][g + ((lane >> 4) << 3)] = dacc[g];
    }
    __syncthreads();

#pragma unroll
    for (int e = 0; e < 16; e++) {
        int idx = e * 128 + tid;
        int Mr  = idx >> 7;
        int n   = idx & 127;
        float s = red[0][Mr][n] + red[1][Mr][n] + red[2][Mr][n] + red[3][Mr][n];
        float d = redd[0][Mr] + redd[1][Mr] + redd[2][Mr] + redd[3][Mr];
        __builtin_nontemporal_store(s / d, &out[(size_t)(i0 + Mr) * OUT_DIM + n]);
    }
}

// ---------------------------------------------------------------------------
extern "C" void kernel_launch(void* const* d_in, const int* in_sizes, int n_in,
                              void* d_out, int out_size, void* d_ws, size_t ws_size,
                              hipStream_t stream) {
    const float* x   = (const float*)d_in[0];
    const int*   adj = (const int*)d_in[1];
    const float* W   = (const float*)d_in[2];
    const float* a   = (const float*)d_in[3];
    float* out = (float*)d_out;

    char* ws = (char*)d_ws;
    float*          Wh  = (float*)ws;                                   // 4 MB
    unsigned short* WhB = (unsigned short*)(ws + 4u * 1024 * 1024);     // 2 MB
    float*          f1  = (float*)(ws + 6u * 1024 * 1024);              // 32 KB
    float*          f2  = (float*)(ws + 6u * 1024 * 1024 + 32 * 1024);  // 32 KB
    float*          mf2 = (float*)(ws + 6u * 1024 * 1024 + 64 * 1024);  // 4 B

    wh_kernel<<<N_NODES, 128, 0, stream>>>(x, W, Wh, WhB);
    f_kernel<<<N_NODES / 256, 256, 0, stream>>>(Wh, a, f1, f2);
    maxf2_kernel<<<1, 256, 0, stream>>>(f2, mf2);
    gat_kernel<<<N_NODES / 16, 128, 0, stream>>>(adj, WhB, f1, f2, mf2, out);
}
